// SpatialAwareAttention_76605036691563
// MI455X (gfx1250) — compile-verified, matrix operand order checked by reading, not run on hardware
//
#include <hip/hip_runtime.h>
#include <hip/hip_bf16.h>

// ---------------------------------------------------------------------------
// SpatialAwareAttention for MI455X (gfx1250, wave32, WMMA 16x16x32 bf16)
// ---------------------------------------------------------------------------
#define B_SZ 4
#define L_SEQ 1024
#define D_MODEL 768
#define N_HEADS 12
#define HEAD_DIM 64
#define BIAS_HID 32

typedef __attribute__((ext_vector_type(16))) __bf16 v16bf;
typedef __attribute__((ext_vector_type(8)))  float  v8f;
typedef __attribute__((ext_vector_type(4)))  unsigned int u32x4;

union FragBF { v16bf v; u32x4 u[2]; };

// CDNA5 async global->LDS DMA (ASYNCcnt-tracked), guarded so we fall back to
// plain load/store staging if this toolchain lacks the builtins.
#if defined(__AMDGCN__) && \
    __has_builtin(__builtin_amdgcn_global_load_async_to_lds_b128) && \
    __has_builtin(__builtin_amdgcn_s_wait_asynccnt)
#define HAVE_ASYNC_LDS 1
#endif

__device__ __forceinline__ void async_copy16(const __bf16* g, __bf16* l) {
#ifdef HAVE_ASYNC_LDS
    // Builtin expects: (addrspace(1) int4*, addrspace(3) int4*, imm off, imm cpol).
    // Build the addrspace pointers via integer casts: the generic LDS pointer's
    // low 32 bits are the LDS offset; the generic global pointer is the full VA.
    typedef int v4i __attribute__((ext_vector_type(4)));
    typedef __attribute__((address_space(1))) v4i* gp_t;
    typedef __attribute__((address_space(3))) v4i* lp_t;
    gp_t gp = (gp_t)(unsigned long long)g;
    lp_t lp = (lp_t)(unsigned int)(unsigned long long)l;
    __builtin_amdgcn_global_load_async_to_lds_b128(gp, lp, 0, 0);
#else
    *(u32x4*)l = *(const u32x4*)g;
#endif
}

__device__ __forceinline__ void async_wait_all() {
#ifdef HAVE_ASYNC_LDS
    __builtin_amdgcn_s_wait_asynccnt(0);
#endif
}

__device__ __forceinline__ v8f wmma_bf16(v16bf a, v16bf b, v8f c) {
    return __builtin_amdgcn_wmma_f32_16x16x32_bf16(false, a, false, b,
                                                   (short)0, c, false, false);
}

// Load a 16x32 bf16 A/B fragment from a row-major [16][ld] array.
// Per ISA 7.12.2: lanes 0-15 hold row (lane), K 0..7 & 16..23;
// lanes 16-31 hold row (lane-16), K 8..15 & 24..31.
__device__ __forceinline__ v16bf frag16x32(const __bf16* base, int ld) {
    const int l = threadIdx.x & 31;
    const __bf16* p = base + (l & 15) * ld + ((l >> 4) << 3);
    FragBF f;
    f.u[0] = *(const u32x4*)(p);
    f.u[1] = *(const u32x4*)(p + 16);
    return f.v;
}

// ---------------------------------------------------------------------------
// Pack / small kernels
// ---------------------------------------------------------------------------
__global__ void cvt_f32_bf16(const float* __restrict__ X, __bf16* __restrict__ Y, int n) {
    int i = blockIdx.x * blockDim.x + threadIdx.x;
    if (i < n) Y[i] = (__bf16)X[i];
}

// Wt[n][k] = W[k][n]  (768x768)
__global__ void transpose_w(const float* __restrict__ W, __bf16* __restrict__ Wt) {
    int i = blockIdx.x * blockDim.x + threadIdx.x;
    if (i >= D_MODEL * D_MODEL) return;
    int n = i / D_MODEL, k = i - n * D_MODEL;
    Wt[n * D_MODEL + k] = (__bf16)W[k * D_MODEL + n];
}

// W2t[head][h] padded to 16 heads, bf16
__global__ void pack_w2(const float* __restrict__ W2, __bf16* __restrict__ W2t) {
    int t = threadIdx.x;            // 512 threads
    int head = t >> 5, h = t & 31;
    float v = (head < N_HEADS) ? W2[h * N_HEADS + head] : 0.0f;
    W2t[head * 32 + h] = (__bf16)v;
}

// a[b,l,:] = spatial_pos[b,l,:] @ W1   (delta@W1 == a_i - a_j)
__global__ void pos_mlp(const float* __restrict__ pos, const float* __restrict__ W1,
                        float* __restrict__ a) {
    int idx = blockIdx.x * blockDim.x + threadIdx.x;   // 0..4095
    if (idx >= B_SZ * L_SEQ) return;
    float px = pos[idx * 2 + 0], py = pos[idx * 2 + 1];
    #pragma unroll
    for (int h = 0; h < BIAS_HID; ++h)
        a[idx * BIAS_HID + h] = px * W1[h] + py * W1[BIAS_HID + h];
}

// ---------------------------------------------------------------------------
// Tiled WMMA GEMM: C[4096,768] = A[4096,768] @ Bt[768,768]^T + bias
// mode 0/1: scatter bf16 to [B,H,L,hd] (Q/K) ; mode 2: scatter bf16 to
// [B,H,hd,L] (V transposed) ; mode 3: f32 row-major (final output)
// Block: 256 threads = 8 waves (4M x 2N of 32x32 wave tiles) -> 128x64 tile.
// Double-buffered LDS with async global->LDS DMA of tile k+1 overlapping the
// WMMAs of tile k.
// ---------------------------------------------------------------------------
__global__ __launch_bounds__(256) void gemm_bf16(const __bf16* __restrict__ A,
                                                 const __bf16* __restrict__ Bt,
                                                 const float*  __restrict__ bias,
                                                 void* __restrict__ outp, int mode) {
    __shared__ __bf16 As[2][128 * 40];   // padded stride 40 elems (80B) for banks
    __shared__ __bf16 Bs[2][64 * 40];

    const int tid  = threadIdx.x;
    const int w    = tid >> 5;
    const int lane = tid & 31;
    const int wm   = w & 3, wn = w >> 2;
    const int m0   = blockIdx.x * 128;
    const int n0   = blockIdx.y * 64;

    auto load_tile = [&](int k0, int p) {
        // A tile 128x32 (2 x 16B/thread), B tile 64x32 (1 x 16B/thread)
        #pragma unroll
        for (int it = 0; it < 2; ++it) {
            int c = tid + it * 256;
            int row = c >> 2, q = c & 3;
            async_copy16(A + (size_t)(m0 + row) * D_MODEL + k0 + q * 8,
                         &As[p][row * 40 + q * 8]);
        }
        int row = tid >> 2, q = tid & 3;
        async_copy16(Bt + (size_t)(n0 + row) * D_MODEL + k0 + q * 8,
                     &Bs[p][row * 40 + q * 8]);
    };

    v8f acc[2][2];
    #pragma unroll
    for (int a = 0; a < 2; ++a)
        #pragma unroll
        for (int b = 0; b < 2; ++b)
            #pragma unroll
            for (int e = 0; e < 8; ++e) acc[a][b][e] = 0.0f;

    load_tile(0, 0);
    async_wait_all();
    __syncthreads();

    int p = 0;
    for (int k0 = 0; k0 < D_MODEL; k0 += 32) {
        if (k0 + 32 < D_MODEL) load_tile(k0 + 32, p ^ 1);   // DMA next tile

        v16bf af0 = frag16x32(&As[p][(wm * 32 + 0)  * 40], 40);
        v16bf af1 = frag16x32(&As[p][(wm * 32 + 16) * 40], 40);
        v16bf bf0 = frag16x32(&Bs[p][(wn * 32 + 0)  * 40], 40);
        v16bf bf1 = frag16x32(&Bs[p][(wn * 32 + 16) * 40], 40);

        acc[0][0] = wmma_bf16(af0, bf0, acc[0][0]);
        acc[0][1] = wmma_bf16(af0, bf1, acc[0][1]);
        acc[1][0] = wmma_bf16(af1, bf0, acc[1][0]);
        acc[1][1] = wmma_bf16(af1, bf1, acc[1][1]);

        async_wait_all();     // next tile landed in LDS
        __syncthreads();
        p ^= 1;
    }

    const int half8 = (lane >> 4) << 3;
    const int nl    = lane & 15;
    #pragma unroll
    for (int a = 0; a < 2; ++a)
        #pragma unroll
        for (int b = 0; b < 2; ++b)
            #pragma unroll
            for (int r = 0; r < 8; ++r) {
                int m = m0 + wm * 32 + a * 16 + r + half8;
                int n = n0 + wn * 32 + b * 16 + nl;
                float val = acc[a][b][r] + bias[n];
                int bb = m >> 10, i = m & 1023;
                int hh = n >> 6,  d = n & 63;
                if (mode <= 1) {           // Q/K -> [B,H,L,hd] bf16
                    ((__bf16*)outp)[((((size_t)(bb * N_HEADS + hh)) << 10) + i) * 64 + d] =
                        (__bf16)val;
                } else if (mode == 2) {    // V -> [B,H,hd,L] bf16 (transposed)
                    ((__bf16*)outp)[((size_t)(bb * N_HEADS + hh) * 64 + d) * 1024 + i] =
                        (__bf16)val;
                } else {                   // final f32 output
                    ((float*)outp)[(size_t)m * D_MODEL + n] = val;
                }
            }
}

// ---------------------------------------------------------------------------
// Fused flash attention with spatial-bias MLP.
// Grid: (L/16, B). Block: 384 threads = 12 waves, wave w owns head w.
// Query tile: 16 rows. Key tiles: 32 columns per step.
// Per step: hidden=relu(aI-aJ) (VALU, shared by all heads) -> LDS;
//           bias = hidden @ W2  via WMMA (32 M-tiles split over 12 waves);
//           S = Q K^T (WMMA) * scale + bias + b2; online softmax;
//           O += P @ V (WMMA, V pre-transposed).
// ---------------------------------------------------------------------------
__global__ __launch_bounds__(384) void attn_fused(
        const __bf16* __restrict__ Qb, const __bf16* __restrict__ Kb,
        const __bf16* __restrict__ Vt, const float* __restrict__ aPre,
        const float* __restrict__ b1,  const float* __restrict__ b2,
        const __bf16* __restrict__ W2t, __bf16* __restrict__ Obf) {
    extern __shared__ char smem[];
    __bf16* hidS  = (__bf16*)(smem);            // [512][40] bf16   40960B
    float*  biasS = (float*)(smem + 40960);     // [512][17] f32    34816B
    float*  aI    = (float*)(smem + 75776);     // [16][32] f32      2048B
    float*  aJ    = (float*)(smem + 77824);     // [32][32] f32      4096B
    __bf16* pSt   = (__bf16*)(smem + 81920);    // [12][16][40] bf16 15360B
                                                // total 97280B

    const int b    = blockIdx.y;
    const int i0   = blockIdx.x << 4;
    const int tid  = threadIdx.x;
    const int w    = tid >> 5;       // wave id == head id (12 waves)
    const int lane = tid & 31;
    const int h    = w;
    const int half8 = (lane >> 4) << 3;
    const int nl    = lane & 15;

    // aI = a[i-rows] + b1 (fold MLP bias into the query side)
    for (int t = tid; t < 16 * 32; t += 384) {
        int r = t >> 5, c = t & 31;
        aI[t] = aPre[((size_t)(b << 10) + i0 + r) * 32 + c] + b1[c];
    }

    const v16bf w2f = frag16x32(W2t, 32);

    const __bf16* qbase = Qb + (((size_t)(b * N_HEADS + h) << 10) + i0) * 64;
    const v16bf qf0 = frag16x32(qbase, 64);
    const v16bf qf1 = frag16x32(qbase + 32, 64);

    const __bf16* kbaseH = Kb + ((size_t)(b * N_HEADS + h) << 10) * 64;
    const __bf16* vbaseH = Vt + (size_t)(b * N_HEADS + h) * 64 * 1024;
    const float b2h = b2[h];

    v8f o[4];
    float mrun[8], lrun[8];
    #pragma unroll
    for (int d = 0; d < 4; ++d)
        #pragma unroll
        for (int e = 0; e < 8; ++e) o[d][e] = 0.0f;
    #pragma unroll
    for (int r = 0; r < 8; ++r) { mrun[r] = -3.0e38f; lrun[r] = 0.0f; }

    for (int j0 = 0; j0 < L_SEQ; j0 += 32) {
        __syncthreads();   // previous-iteration readers of hidS/biasS/aJ done
        // load aJ tile
        for (int t = tid; t < 32 * 32; t += 384) {
            int r = t >> 5, c = t & 31;
            aJ[t] = aPre[((size_t)(b << 10) + j0 + r) * 32 + c];
        }
        __syncthreads();

        // hidden = relu(aI - aJ)  for 512 (i,j) pairs, shared by all heads
        for (int p = tid; p < 512; p += 384) {
            const float* ai = &aI[(p >> 5) * 32];
            const float* aj = &aJ[(p & 31) * 32];
            __bf16* hp = &hidS[p * 40];
            #pragma unroll
            for (int c = 0; c < BIAS_HID; ++c)
                hp[c] = (__bf16)fmaxf(ai[c] - aj[c], 0.0f);
        }
        __syncthreads();

        // bias = hidden @ W2  (32 WMMA M-tiles over 12 waves)
        for (int t = w; t < 32; t += 12) {
            v16bf hf = frag16x32(&hidS[t * 16 * 40], 40);
            v8f c;
            #pragma unroll
            for (int e = 0; e < 8; ++e) c[e] = 0.0f;
            c = wmma_bf16(hf, w2f, c);
            #pragma unroll
            for (int r = 0; r < 8; ++r)
                biasS[(t * 16 + r + half8) * 17 + nl] = c[r];
        }
        __syncthreads();

        // prefetch next K/V tiles into L2 (global_prefetch_b8)
        if (j0 + 32 < L_SEQ) {
            __builtin_prefetch(kbaseH + (size_t)(j0 + 32 + lane) * 64, 0, 3);
            __builtin_prefetch(vbaseH + (size_t)lane * 1024 + j0 + 32, 0, 3);
            __builtin_prefetch(vbaseH + (size_t)(lane + 32) * 1024 + j0 + 32, 0, 3);
        }

        // S = Q K^T  (two 16x16 subtiles, K=64 -> 2 WMMAs each)
        const __bf16* kb = kbaseH + (size_t)j0 * 64;
        v8f s[2];
        #pragma unroll
        for (int ns = 0; ns < 2; ++ns) {
            v8f c;
            #pragma unroll
            for (int e = 0; e < 8; ++e) c[e] = 0.0f;
            c = wmma_bf16(qf0, frag16x32(kb + ns * 16 * 64, 64), c);
            c = wmma_bf16(qf1, frag16x32(kb + ns * 16 * 64 + 32, 64), c);
            s[ns] = c;
        }
        // scale + spatial bias + b2
        #pragma unroll
        for (int ns = 0; ns < 2; ++ns)
            #pragma unroll
            for (int r = 0; r < 8; ++r) {
                int p = ((r + half8) << 5) | (ns * 16 + nl);
                s[ns][r] = s[ns][r] * 0.125f + biasS[p * 17 + h] + b2h;
            }

        // online softmax
        float mnew[8], alpha[8], rsum[8];
        #pragma unroll
        for (int r = 0; r < 8; ++r) {
            float mt = fmaxf(s[0][r], s[1][r]);
            #pragma unroll
            for (int msk = 1; msk < 16; msk <<= 1)
                mt = fmaxf(mt, __shfl_xor(mt, msk, 32));
            mnew[r]  = fmaxf(mrun[r], mt);
            alpha[r] = __expf(mrun[r] - mnew[r]);
            mrun[r]  = mnew[r];
        }
        #pragma unroll
        for (int r = 0; r < 8; ++r) {
            s[0][r] = __expf(s[0][r] - mnew[r]);
            s[1][r] = __expf(s[1][r] - mnew[r]);
            float rs = s[0][r] + s[1][r];
            #pragma unroll
            for (int msk = 1; msk < 16; msk <<= 1)
                rs += __shfl_xor(rs, msk, 32);
            rsum[r] = rs;
        }
        #pragma unroll
        for (int r = 0; r < 8; ++r) {
            lrun[r] = lrun[r] * alpha[r] + rsum[r];
            #pragma unroll
            for (int d = 0; d < 4; ++d) o[d][r] *= alpha[r];
        }

        // stage P (16x32 bf16) in per-wave LDS, reload as A-fragment
        __bf16* pw = pSt + w * 16 * 40;
        #pragma unroll
        for (int ns = 0; ns < 2; ++ns)
            #pragma unroll
            for (int r = 0; r < 8; ++r)
                pw[(r + half8) * 40 + ns * 16 + nl] = (__bf16)s[ns][r];
        v16bf pf = frag16x32(pw, 40);

        // O += P @ V   (V pre-transposed: rows = hd, contiguous in j)
        #pragma unroll
        for (int d = 0; d < 4; ++d)
            o[d] = wmma_bf16(pf, frag16x32(vbaseH + (size_t)(d * 16) * 1024 + j0, 1024),
                             o[d]);
    }

    // epilogue: normalize and store bf16 row-major [B*L, 768]
    #pragma unroll
    for (int r = 0; r < 8; ++r) {
        float inv = 1.0f / lrun[r];
        int row = i0 + r + half8;
        #pragma unroll
        for (int d = 0; d < 4; ++d) {
            int col = h * 64 + d * 16 + nl;
            Obf[(size_t)((b << 10) + row) * D_MODEL + col] = (__bf16)(o[d][r] * inv);
        }
    }
}

// ---------------------------------------------------------------------------
// Host launcher
// ---------------------------------------------------------------------------
extern "C" void kernel_launch(void* const* d_in, const int* in_sizes, int n_in,
                              void* d_out, int out_size, void* d_ws, size_t ws_size,
                              hipStream_t stream) {
    const float* x   = (const float*)d_in[0];
    const float* pos = (const float*)d_in[1];
    const float* Wq  = (const float*)d_in[2];
    const float* bq  = (const float*)d_in[3];
    const float* Wk  = (const float*)d_in[4];
    const float* bk  = (const float*)d_in[5];
    const float* Wv  = (const float*)d_in[6];
    const float* bv  = (const float*)d_in[7];
    const float* Wo  = (const float*)d_in[8];
    const float* bo  = (const float*)d_in[9];
    const float* W1  = (const float*)d_in[10];
    const float* b1  = (const float*)d_in[11];
    const float* W2  = (const float*)d_in[12];
    const float* b2  = (const float*)d_in[13];

    const size_t ML   = (size_t)B_SZ * L_SEQ;          // 4096
    const size_t XBE  = ML * D_MODEL;                  // 3145728 elements
    const size_t WBE  = (size_t)D_MODEL * D_MODEL;     // 589824 elements
    const size_t QKVE = (size_t)B_SZ * N_HEADS * L_SEQ * HEAD_DIM;

    char* ws = (char*)d_ws;
    size_t off = 0;
    auto take = [&](size_t bytes) -> char* {
        char* p = ws + off;
        off += (bytes + 255) & ~(size_t)255;
        return p;
    };
    __bf16* xb  = (__bf16*)take(XBE * 2);
    __bf16* Wqt = (__bf16*)take(WBE * 2);
    __bf16* Wkt = (__bf16*)take(WBE * 2);
    __bf16* Wvt = (__bf16*)take(WBE * 2);
    __bf16* Wot = (__bf16*)take(WBE * 2);
    __bf16* Qb  = (__bf16*)take(QKVE * 2);
    __bf16* Kb  = (__bf16*)take(QKVE * 2);
    __bf16* Vt  = (__bf16*)take(QKVE * 2);
    __bf16* Obf = (__bf16*)take(XBE * 2);
    float*  aP  = (float*)take(ML * BIAS_HID * 4);
    __bf16* W2t = (__bf16*)take(16 * 32 * 2);

    // 1. packing
    cvt_f32_bf16<<<dim3((unsigned)((XBE + 255) / 256)), 256, 0, stream>>>(x, xb, (int)XBE);
    const unsigned wtg = (unsigned)((WBE + 255) / 256);
    transpose_w<<<wtg, 256, 0, stream>>>(Wq, Wqt);
    transpose_w<<<wtg, 256, 0, stream>>>(Wk, Wkt);
    transpose_w<<<wtg, 256, 0, stream>>>(Wv, Wvt);
    transpose_w<<<wtg, 256, 0, stream>>>(Wo, Wot);
    pack_w2<<<1, 512, 0, stream>>>(W2, W2t);
    pos_mlp<<<(unsigned)((ML + 255) / 256), 256, 0, stream>>>(pos, W1, aP);

    // 2. Q/K/V projections (WMMA)
    dim3 ggrid(32, 12, 1);
    gemm_bf16<<<ggrid, 256, 0, stream>>>(xb, Wqt, bq, (void*)Qb, 0);
    gemm_bf16<<<ggrid, 256, 0, stream>>>(xb, Wkt, bk, (void*)Kb, 1);
    gemm_bf16<<<ggrid, 256, 0, stream>>>(xb, Wvt, bv, (void*)Vt, 2);

    // 3. fused flash attention + spatial bias MLP
    const size_t smem = 97280;
    attn_fused<<<dim3(L_SEQ / 16, B_SZ, 1), 384, smem, stream>>>(
        Qb, Kb, Vt, aP, b1, b2, W2t, Obf);

    // 4. output projection (WMMA, f32 out)
    gemm_bf16<<<ggrid, 256, 0, stream>>>(Obf, Wot, bo, d_out, 3);
}
